// MambaBlock_15161234555020
// MI455X (gfx1250) — matrix operand order repaired and checked by reading, not compiled, not run on hardware
//
#include <hip/hip_runtime.h>
#include <hip/hip_bf16.h>
#include <stdint.h>

// ---------------------------------------------------------------- types
typedef __bf16 bf16_t;
typedef __attribute__((ext_vector_type(16))) __bf16 v16bf;
typedef __attribute__((ext_vector_type(8)))  __bf16 v8bf;
typedef __attribute__((ext_vector_type(8)))  float  v8f;

// ---------------------------------------------------------------- shapes
#define B_    4
#define L_    1024
#define D_    256
#define N_ST  16
#define DTR_  16
#define KCONV 4
#define ML    (B_ * L_)         // 4096 rows
#define XDBL  (DTR_ + 2 * N_ST) // 48

// ---------------------------------------------------------------- helpers
__global__ void cvt_f32_to_bf16_kernel(const float* __restrict__ src,
                                       bf16_t* __restrict__ dst, int n) {
  int i = blockIdx.x * blockDim.x + threadIdx.x;
  if (i < n) dst[i] = (bf16_t)src[i];
}

// disT[b, s, l] = bf16(dis[b, l, s]) -- 32x32 LDS-tiled transpose so both the
// fp32 reads and the bf16 writes are coalesced.
__global__ void transpose_dis_kernel(const float* __restrict__ dis,
                                     bf16_t* __restrict__ disT) {
  __shared__ bf16_t tile[32][33];
  int b  = blockIdx.z;
  int l0 = blockIdx.y * 32;  // source row block (l)
  int s0 = blockIdx.x * 32;  // source col block (s)
  int tx = threadIdx.x;      // 0..31
  int ty = threadIdx.y;      // 0..7
#pragma unroll
  for (int j = 0; j < 4; ++j) {
    int ly = ty + j * 8;
    tile[ly][tx] = (bf16_t)dis[(size_t)b * L_ * L_ + (size_t)(l0 + ly) * L_ + s0 + tx];
  }
  __syncthreads();
#pragma unroll
  for (int j = 0; j < 4; ++j) {
    int sy = ty + j * 8;
    disT[(size_t)b * L_ * L_ + (size_t)(s0 + sy) * L_ + l0 + tx] = tile[tx][sy];
  }
}

// ---------------------------------------------------------------- WMMA GEMM
// C[M,N] = A[M,K] * Bsrc[N,K]^T  (bf16 operands, fp32 accumulate).
// One 16 x (16*NB) output tile per wave32: the A fragment is loaded once per
// K-step and reused across NB B fragments / NB accumulators.
// A fragment: lane m = lane&15, K halves 0-7 / +16 per lane-half (v8bf x2).
// B fragment: lane n = lane&15, K halves 0-7 / +16 per lane-half (v8bf x2).
template <int NB>
__global__ void wmma_gemm_bf16_kernel(const bf16_t* __restrict__ A,
                                      const bf16_t* __restrict__ Bsrc,
                                      float* __restrict__ C,
                                      int M, int N, int K,
                                      long long strideA, long long strideB,
                                      long long strideC, int finiteMask) {
  int bz = blockIdx.y;
  A    += (size_t)bz * strideA;
  Bsrc += (size_t)bz * strideB;
  C    += (size_t)bz * strideC;

  int wave = (blockIdx.x * blockDim.x + threadIdx.x) >> 5;
  int lane = threadIdx.x & 31;
  int tilesM  = M >> 4;
  int tilesNb = N / (16 * NB);
  if (wave >= tilesM * tilesNb) return;  // whole-wave guard: EXEC stays all-1s

  int m0 = (wave / tilesNb) << 4;
  int n0 = (wave % tilesNb) * (16 * NB);
  int half = lane >> 4;   // K-half select
  int mr   = lane & 15;   // row (A) / col (B) within fragment

  const bf16_t* arow = A + (size_t)(m0 + mr) * K + half * 8;
  const bf16_t* brow = Bsrc + (size_t)(n0 + mr) * K + half * 8;

  v8f zero = {0.f, 0.f, 0.f, 0.f, 0.f, 0.f, 0.f, 0.f};
  v8f acc[NB];
#pragma unroll
  for (int f = 0; f < NB; ++f) acc[f] = zero;

  for (int k0 = 0; k0 < K; k0 += 32) {
    __builtin_prefetch(arow + k0 + 64, 0, 1);
    __builtin_prefetch(brow + k0 + 64, 0, 1);
    v8bf a_lo = *(const v8bf*)(arow + k0);
    v8bf a_hi = *(const v8bf*)(arow + k0 + 16);
    v16bf av = __builtin_shufflevector(a_lo, a_hi, 0, 1, 2, 3, 4, 5, 6, 7,
                                       8, 9, 10, 11, 12, 13, 14, 15);
    v16bf bv[NB];
#pragma unroll
    for (int f = 0; f < NB; ++f) {
      const bf16_t* bf = brow + (size_t)f * 16 * K + k0;
      v8bf b_lo = *(const v8bf*)(bf);
      v8bf b_hi = *(const v8bf*)(bf + 16);
      bv[f] = __builtin_shufflevector(b_lo, b_hi, 0, 1, 2, 3, 4, 5, 6, 7,
                                      8, 9, 10, 11, 12, 13, 14, 15);
    }
#pragma unroll
    for (int f = 0; f < NB; ++f) {
      acc[f] = __builtin_amdgcn_wmma_f32_16x16x32_bf16(
          /*neg_a=*/false, av, /*neg_b=*/false, bv[f],
          /*c_mod=*/(short)0, acc[f], /*reuse_a=*/false, /*reuse_b=*/false);
    }
  }

  // D layout: lanes 0-15 -> M = r, lanes 16-31 -> M = 8 + r ; N = lane&15
#pragma unroll
  for (int f = 0; f < NB; ++f) {
#pragma unroll
    for (int r = 0; r < 8; ++r) {
      int row = m0 + r + 8 * half;
      float v = acc[f][r];
      if (finiteMask) v = __builtin_isfinite(v) ? v : 0.0f;
      C[(size_t)row * N + n0 + f * 16 + mr] = v;
    }
  }
}

// ---------------------------------------------------------------- conv+SiLU
// u = silu(depthwise_causal_conv(xc) + b); also emit bf16 u and bf16 u^T.
__global__ void conv_silu_kernel(const float* __restrict__ xand,   // (ML, 2D)
                                 const float* __restrict__ conv_w, // (D,1,K)
                                 const float* __restrict__ conv_b, // (D)
                                 float* __restrict__ u,            // (ML, D)
                                 bf16_t* __restrict__ u_bf,        // (ML, D)
                                 bf16_t* __restrict__ uT_bf) {     // (B, D, L)
  int i = blockIdx.x * blockDim.x + threadIdx.x;  // B*L*D
  if (i >= B_ * L_ * D_) return;
  int d = i % D_;
  int t = i / D_;
  int l = t % L_;
  int b = t / L_;
  float acc = conv_b[d];
#pragma unroll
  for (int k = 0; k < KCONV; ++k) {
    int ls = l - (KCONV - 1) + k;
    if (ls >= 0)
      acc += xand[((size_t)b * L_ + ls) * (2 * D_) + d] * conv_w[d * KCONV + k];
  }
  float s = acc / (1.0f + __expf(-acc));  // silu
  u[i] = s;
  u_bf[i] = (bf16_t)s;
  uT_bf[((size_t)b * D_ + d) * L_ + l] = (bf16_t)s;
}

// ---------------------------------------------------------------- scan
// 16 lanes (n = lane&15) cooperate per (b,d) channel. Keeps the reference's
// clipped log-cumsum semantics exactly; reduces over N with wave32 shuffles.
// Writes the gated output y*silu(res) in bf16 (A operand of out-proj GEMM).
__global__ void scan_gate_kernel(const float* __restrict__ delta_p, // (B,L,D)
                                 const float* __restrict__ u,       // (ML,D)
                                 const float* __restrict__ x_dbl,   // (ML,48)
                                 const float* __restrict__ xand,    // (ML,2D)
                                 const float* __restrict__ A_log,   // (D,N)
                                 const float* __restrict__ D_param, // (D)
                                 bf16_t* __restrict__ ygated_bf) {  // (ML,D)
  int g = blockIdx.x * blockDim.x + threadIdx.x;  // B*D*16
  if (g >= B_ * D_ * N_ST) return;
  int n  = g & 15;
  int bd = g >> 4;
  int d  = bd % D_;
  int b  = bd / D_;

  float An = -__expf(A_log[d * N_ST + n]);  // A = -exp(A_log)
  float Dp = D_param[d];
  float run = 0.0f;  // unclipped cumsum of clipped log_deltaA
  float c   = 0.0f;  // cumsum of deltaBu * exp(-log_P)

  for (int l = 0; l < L_; ++l) {
    size_t row = (size_t)b * L_ + l;
    float dp = delta_p[row * D_ + d];
    float uu = u[row * D_ + d];
    float la = fminf(fmaxf(dp * An, -20.0f), 20.0f);
    run += la;
    float lp = fminf(fmaxf(run, -30.0f), 30.0f);
    float Bn = x_dbl[row * XDBL + DTR_ + n];
    float Cn = x_dbl[row * XDBL + DTR_ + N_ST + n];
    float elp = __expf(lp);                       // exp(log_P)
    c += dp * Bn * uu * __builtin_amdgcn_rcpf(elp);  // * exp(-log_P)
    float yn = c * elp * Cn;
    // sum over the 16 state lanes
    yn += __shfl_xor(yn, 8, 16);
    yn += __shfl_xor(yn, 4, 16);
    yn += __shfl_xor(yn, 2, 16);
    yn += __shfl_xor(yn, 1, 16);
    if (n == 0) {
      float y = yn + uu * Dp;
      float r = xand[row * (2 * D_) + D_ + d];  // res
      float sr = r / (1.0f + __expf(-r));       // silu(res)
      ygated_bf[row * D_ + d] = (bf16_t)(y * sr);
    }
  }
}

// ---------------------------------------------------------------- launcher
extern "C" void kernel_launch(void* const* d_in, const int* in_sizes, int n_in,
                              void* d_out, int out_size, void* d_ws,
                              size_t ws_size, hipStream_t stream) {
  (void)in_sizes; (void)n_in; (void)out_size; (void)ws_size;

  const float* x       = (const float*)d_in[0];   // (B,L,D)
  const float* dis     = (const float*)d_in[1];   // (B,L,L)
  const float* W_in    = (const float*)d_in[2];   // (2D,D)
  const float* conv_w  = (const float*)d_in[3];   // (D,1,K)
  const float* conv_b  = (const float*)d_in[4];   // (D)
  const float* W_x     = (const float*)d_in[5];   // (48,D)
  // d_in[6] W_dt, d_in[7] b_dt: dead code in reference (delta unused)
  const float* A_log   = (const float*)d_in[8];   // (D,N)
  const float* D_param = (const float*)d_in[9];   // (D)
  const float* W_out   = (const float*)d_in[10];  // (D,D)
  float* out = (float*)d_out;

  // ---- workspace layout (256B-aligned regions)
  uint8_t* ws = (uint8_t*)d_ws;
  size_t off = 0;
  auto take = [&](size_t bytes) {
    size_t r = off;
    off += (bytes + 255) & ~(size_t)255;
    return r;
  };
  float*  x_and_res = (float*)(ws + take((size_t)ML * 2 * D_ * 4));   // 8 MB
  float*  u_f32     = (float*)(ws + take((size_t)ML * D_ * 4));       // 4 MB
  float*  delta_p   = (float*)(ws + take((size_t)ML * D_ * 4));       // 4 MB
  float*  x_dbl     = (float*)(ws + take((size_t)ML * XDBL * 4));     // .75 MB
  bf16_t* X_bf      = (bf16_t*)(ws + take((size_t)ML * D_ * 2));      // 2 MB
  bf16_t* Win_bf    = (bf16_t*)(ws + take((size_t)2 * D_ * D_ * 2));  // .25 MB
  bf16_t* u_bf      = (bf16_t*)(ws + take((size_t)ML * D_ * 2));      // 2 MB
  bf16_t* uT_bf     = (bf16_t*)(ws + take((size_t)B_ * D_ * L_ * 2)); // 2 MB
  bf16_t* Wx_bf     = (bf16_t*)(ws + take((size_t)XDBL * D_ * 2));
  bf16_t* disT_bf   = (bf16_t*)(ws + take((size_t)B_ * L_ * L_ * 2)); // 8 MB
  bf16_t* yg_bf     = (bf16_t*)(ws + take((size_t)ML * D_ * 2));      // 2 MB
  bf16_t* Wout_bf   = (bf16_t*)(ws + take((size_t)D_ * D_ * 2));

  // ---- stage bf16 operands
  {
    int n = ML * D_;
    cvt_f32_to_bf16_kernel<<<(n + 255) / 256, 256, 0, stream>>>(x, X_bf, n);
    n = 2 * D_ * D_;
    cvt_f32_to_bf16_kernel<<<(n + 255) / 256, 256, 0, stream>>>(W_in, Win_bf, n);
    n = XDBL * D_;
    cvt_f32_to_bf16_kernel<<<(n + 255) / 256, 256, 0, stream>>>(W_x, Wx_bf, n);
    n = D_ * D_;
    cvt_f32_to_bf16_kernel<<<(n + 255) / 256, 256, 0, stream>>>(W_out, Wout_bf, n);
    transpose_dis_kernel<<<dim3(L_ / 32, L_ / 32, B_), dim3(32, 8), 0, stream>>>(
        dis, disT_bf);
  }

  auto gemm_blocks = [](int waves) { return (waves + 3) / 4; };  // 4 waves/blk

  // ---- 1) x_and_res = X * W_in^T : (4096,256)x(512,256)^T -> (4096,512)
  {
    int waves = (ML / 16) * ((2 * D_) / 64);  // 2048
    wmma_gemm_bf16_kernel<4><<<dim3(gemm_blocks(waves), 1), 128, 0, stream>>>(
        X_bf, Win_bf, x_and_res, ML, 2 * D_, D_, 0, 0, 0, 0);
  }

  // ---- 2) depthwise causal conv + SiLU -> u (fp32 + bf16 + bf16^T)
  {
    int n = B_ * L_ * D_;
    conv_silu_kernel<<<(n + 255) / 256, 256, 0, stream>>>(
        x_and_res, conv_w, conv_b, u_f32, u_bf, uT_bf);
  }

  // ---- 3) x_dbl = u * W_x^T : (4096,256)x(48,256)^T -> (4096,48)
  {
    int waves = (ML / 16) * (XDBL / 48);  // 256, NB=3 covers all 48 cols
    wmma_gemm_bf16_kernel<3><<<dim3(gemm_blocks(waves), 1), 128, 0, stream>>>(
        u_bf, Wx_bf, x_dbl, ML, XDBL, D_, 0, 0, 0, 0);
  }

  // ---- 4) delta_p[b] = dis[b]^T * u[b] : (1024,1024)x(256,1024)^T per batch
  {
    int waves = (L_ / 16) * (D_ / 64);  // 256 per batch
    wmma_gemm_bf16_kernel<4><<<dim3(gemm_blocks(waves), B_), 128, 0, stream>>>(
        disT_bf, uT_bf, delta_p, L_, D_, L_,
        (long long)L_ * L_, (long long)D_ * L_, (long long)L_ * D_, 0);
  }

  // ---- 5) selective scan + D skip + gating -> ygated (bf16)
  {
    int n = B_ * D_ * N_ST;  // 16384 lanes
    scan_gate_kernel<<<(n + 255) / 256, 256, 0, stream>>>(
        delta_p, u_f32, x_dbl, x_and_res, A_log, D_param, yg_bf);
  }

  // ---- 6) out = ygated * W_out^T, with finite mask -> d_out
  {
    int waves = (ML / 16) * (D_ / 64);  // 1024
    wmma_gemm_bf16_kernel<4><<<dim3(gemm_blocks(waves), 1), 128, 0, stream>>>(
        yg_bf, Wout_bf, out, ML, D_, D_, 0, 0, 0, 1);
  }
}
